// LLaMABlock_53970559041585
// MI455X (gfx1250) — matrix operand order repaired and checked by reading, not compiled, not run on hardware
//
#include <hip/hip_runtime.h>

// ---------------- model constants ----------------
#define T_LEN 2048
#define EMB   4096
#define NH    32
#define NKVH  8
#define DH    128
#define HID   11008
#define NQKV  6144   // 4096 q + 1024 k + 1024 v
#define NGU   22016  // gate | up

// ---------------- WMMA types ----------------
typedef __attribute__((ext_vector_type(16))) __bf16 bf16x16;
typedef __attribute__((ext_vector_type(8)))  float  f32x8;

union AFrag { bf16x16 v; unsigned int u[8]; };

__device__ __forceinline__ __bf16 to_bf16(float f) {
    union { float f; unsigned int u; } a; a.f = f;
    unsigned int r = a.u + 0x7FFFu + ((a.u >> 16) & 1u);   // round-to-nearest-even
    union { unsigned short s; __bf16 b; } c;
    c.s = (unsigned short)(r >> 16);
    return c.b;
}

__device__ __forceinline__ f32x8 zero8() {
    f32x8 z;
    #pragma unroll
    for (int i = 0; i < 8; ++i) z[i] = 0.f;
    return z;
}

// Async DMA: 16 bytes per lane, global -> LDS, tracked by ASYNCcnt.
// VDST = LDS byte address (low 32 bits of the flat pointer per the LDS
// aperture rule), VADDR = 64-bit global address.
__device__ __forceinline__ void async_load_b128(void* lds_ptr, const void* gptr) {
    unsigned lds_addr = (unsigned)(__UINTPTR_TYPE__)lds_ptr;
    asm volatile("global_load_async_to_lds_b128 %0, %1, off"
                 :: "v"(lds_addr), "v"(gptr) : "memory");
}
__device__ __forceinline__ void wait_async_le4() {
    asm volatile("s_wait_asynccnt 4" ::: "memory");
}
__device__ __forceinline__ void wait_async_0() {
    asm volatile("s_wait_asynccnt 0" ::: "memory");
}

// =============================================================
// RMSNorm: fp32 in, bf16 out.  One block per row.
// =============================================================
__global__ __launch_bounds__(256)
void rmsnorm_kernel(const float* __restrict__ x, const float* __restrict__ w,
                    __bf16* __restrict__ out, int cols)
{
    const int row = blockIdx.x;
    const float* xr = x + (size_t)row * cols;
    float s = 0.f;
    for (int i = threadIdx.x; i < cols; i += 256) { float v = xr[i]; s += v * v; }
    __shared__ float red[256];
    red[threadIdx.x] = s;
    __syncthreads();
    for (int off = 128; off > 0; off >>= 1) {
        if ((int)threadIdx.x < off) red[threadIdx.x] += red[threadIdx.x + off];
        __syncthreads();
    }
    const float r = rsqrtf(red[0] / (float)cols + 1e-5f);
    __bf16* orow = out + (size_t)row * cols;
    for (int i = threadIdx.x; i < cols; i += 256)
        orow[i] = to_bf16(xr[i] * r * w[i]);
}

// =============================================================
// fp32 -> bf16 cast, TRANSPOSED placement: src [R][cols] row-major,
// dst[(dstColOff + c) * rowsK + r].  Used once per call to build
// [N][K] bf16 weights so GEMM tiles are contiguous for async DMA.
// =============================================================
__global__ __launch_bounds__(256)
void copy_cast_T_kernel(const float* __restrict__ src, __bf16* __restrict__ dst,
                        long long n, int cols, long long rowsK, long long dstColOff)
{
    long long i = (long long)blockIdx.x * 256 + threadIdx.x;
    if (i >= n) return;
    long long r = i / cols, c = i % cols;
    dst[(dstColOff + c) * rowsK + r] = to_bf16(src[i]);
}

// =============================================================
// Tiled bf16 WMMA GEMM:  C[M,N] = A[M,K] * Bt[N,K]^T (+ resid)
// Block = 256 thr (8 waves).  Tile 128x128, K-step 32.
// Wave grid 2(M) x 4(N); each wave computes 4x2 tiles of 16x16.
// Tiles staged to LDS with global_load_async_to_lds_b128 (ASYNCcnt),
// double-buffered.
// =============================================================
#define TILE_M 128
#define TILE_N 128
#define TILE_K 32

__global__ __launch_bounds__(256)
void gemm_bf16_wmma(const __bf16* __restrict__ A, const __bf16* __restrict__ Bt,
                    const float* __restrict__ resid, float* __restrict__ C,
                    int M, int N, int K)
{
    __shared__ __bf16 sA[2][TILE_M][TILE_K];      // [m][k]
    __shared__ __bf16 sB[2][TILE_N][TILE_K];      // [n][k]

    const int tid    = threadIdx.x;
    const int lane   = tid & 31;
    const int wave   = tid >> 5;
    const int waveM  = wave >> 2;   // 0..1
    const int waveN  = wave & 3;    // 0..3
    const int lane15 = lane & 15;
    const int laneHi = lane >> 4;

    const int n0 = blockIdx.x * TILE_N;
    const int m0 = blockIdx.y * TILE_M;
    const int nk = K / TILE_K;

    f32x8 acc[4][2];
    #pragma unroll
    for (int mt = 0; mt < 4; ++mt)
        #pragma unroll
        for (int nt = 0; nt < 2; ++nt) acc[mt][nt] = zero8();

    // 4 async B128 copies per thread per tile-pair (2 for A, 2 for B):
    // 256 thr x 2 x 16B = 8KB per matrix = one 128x32 bf16 tile.
    auto issueTiles = [&](int kb, int buf) {
        const int k0 = kb * TILE_K;
        #pragma unroll
        for (int i = 0; i < 2; ++i) {
            int l = tid + i * 256;
            int r = l >> 2;            // 0..127
            int c = (l & 3) * 8;       // 0,8,16,24
            async_load_b128(&sA[buf][r][c], &A [(size_t)(m0 + r) * K + k0 + c]);
            async_load_b128(&sB[buf][r][c], &Bt[(size_t)(n0 + r) * K + k0 + c]);
        }
    };

    issueTiles(0, 0);

    for (int kb = 0; kb < nk; ++kb) {
        const int cur = kb & 1;
        if (kb + 1 < nk) {
            issueTiles(kb + 1, cur ^ 1);       // overlap next DMA with compute
            wait_async_le4();                  // current buffer's 4 copies done
        } else {
            wait_async_0();
        }
        if (kb + 2 < nk) {                     // global_prefetch_b8 for kb+2
            __builtin_prefetch(&A [(size_t)(m0 + (tid & 127)) * K + (size_t)(kb + 2) * TILE_K], 0, 0);
            __builtin_prefetch(&Bt[(size_t)(n0 + (tid & 127)) * K + (size_t)(kb + 2) * TILE_K], 0, 0);
        }
        __syncthreads();                       // all waves' DMA visible

        AFrag bfr[2];
        #pragma unroll
        for (int nt = 0; nt < 2; ++nt) {
            const unsigned int* p = reinterpret_cast<const unsigned int*>(
                &sB[cur][waveN * 32 + nt * 16 + lane15][0]);
            #pragma unroll
            for (int j = 0; j < 8; ++j) bfr[nt].u[j] = p[(j >> 2) * 8 + laneHi * 4 + (j & 3)];
        }
        #pragma unroll
        for (int mt = 0; mt < 4; ++mt) {
            AFrag a;
            const unsigned int* p = reinterpret_cast<const unsigned int*>(
                &sA[cur][waveM * 64 + mt * 16 + lane15][0]);
            const int pb = laneHi * 4;
            #pragma unroll
            for (int j = 0; j < 4; ++j) a.u[j]     = p[pb + j];       // K pairs 0-7 / 8-15
            #pragma unroll
            for (int j = 0; j < 4; ++j) a.u[4 + j] = p[8 + pb + j];   // K pairs 16-23 / 24-31
            #pragma unroll
            for (int nt = 0; nt < 2; ++nt)
                acc[mt][nt] = __builtin_amdgcn_wmma_f32_16x16x32_bf16(
                    false, a.v, false, bfr[nt].v, (short)0, acc[mt][nt], false, false);
        }
        __syncthreads();                       // reads done before next DMA overwrites
    }

    #pragma unroll
    for (int mt = 0; mt < 4; ++mt)
        #pragma unroll
        for (int nt = 0; nt < 2; ++nt) {
            const int col = n0 + waveN * 32 + nt * 16 + lane15;
            #pragma unroll
            for (int r = 0; r < 8; ++r) {
                const int row = m0 + waveM * 64 + mt * 16 + r + laneHi * 8;
                const size_t idx = (size_t)row * N + col;
                float v = acc[mt][nt][r];
                if (resid) v += resid[idx];
                C[idx] = v;
            }
        }
}

// =============================================================
// RoPE + layout change.  qkv row = [q 4096 | k 1024 | v 1024] fp32.
// Q -> [NH][T][DH] bf16, K -> [NKVH][T][DH] bf16, V -> [NKVH][DH][T] bf16 (transposed).
// =============================================================
__global__ __launch_bounds__(256)
void rope_kernel(const float* __restrict__ qkv, __bf16* __restrict__ Qb,
                 __bf16* __restrict__ Kb, __bf16* __restrict__ Vt)
{
    const int t = blockIdx.x;
    const float* row = qkv + (size_t)t * NQKV;
    const float kLog = 9.210340371976184f / (float)DH;   // log(10000)/128

    for (int p = threadIdx.x; p < NH * 64; p += 256) {
        int hh = p >> 6, i = p & 63;
        float inv = __expf(-(float)(2 * i) * kLog);
        float sn, cs; __sincosf((float)t * inv, &sn, &cs);
        float x1 = row[hh * DH + 2 * i], x2 = row[hh * DH + 2 * i + 1];
        __bf16* q = Qb + ((size_t)hh * T_LEN + t) * DH;
        q[2 * i]     = to_bf16(x1 * cs - x2 * sn);
        q[2 * i + 1] = to_bf16(x1 * sn + x2 * cs);
    }
    for (int p = threadIdx.x; p < NKVH * 64; p += 256) {
        int hh = p >> 6, i = p & 63;
        float inv = __expf(-(float)(2 * i) * kLog);
        float sn, cs; __sincosf((float)t * inv, &sn, &cs);
        const float* kr = row + EMB;
        float x1 = kr[hh * DH + 2 * i], x2 = kr[hh * DH + 2 * i + 1];
        __bf16* k = Kb + ((size_t)hh * T_LEN + t) * DH;
        k[2 * i]     = to_bf16(x1 * cs - x2 * sn);
        k[2 * i + 1] = to_bf16(x1 * sn + x2 * cs);
    }
    for (int j = threadIdx.x; j < NKVH * DH; j += 256) {
        int hh = j >> 7, d = j & 127;
        Vt[((size_t)hh * DH + d) * T_LEN + t] = to_bf16(row[EMB + NKVH * DH + j]);
    }
}

// =============================================================
// Causal flash attention, one wave per (16-query tile, head).
// Scores and PV both on v_wmma_f32_16x16x32_bf16; 32 keys / iter.
// Output: bf16 [T][NH*DH].
// =============================================================
__global__ __launch_bounds__(32)
void attn_kernel(const __bf16* __restrict__ Qb, const __bf16* __restrict__ Kb,
                 const __bf16* __restrict__ Vt, __bf16* __restrict__ attnb)
{
    __shared__ __bf16 sP[16][32];

    const int lane   = threadIdx.x;
    const int lane15 = lane & 15;
    const int laneHi = lane >> 4;
    const int q0  = blockIdx.x * 16;
    const int h   = blockIdx.y;
    const int kvh = h >> 2;                 // H/KVH = 4
    const float scale = 0.08838834764831845f;   // 1/sqrt(128)

    // Q fragments for the four 32-wide d-chunks (kept in registers)
    AFrag qf[4];
    {
        const unsigned int* qrow = reinterpret_cast<const unsigned int*>(
            &Qb[((size_t)h * T_LEN + q0 + lane15) * DH]);
        const int pb = laneHi * 4;
        #pragma unroll
        for (int c = 0; c < 4; ++c) {
            const unsigned int* p = qrow + c * 16;
            #pragma unroll
            for (int j = 0; j < 4; ++j) qf[c].u[j]     = p[pb + j];
            #pragma unroll
            for (int j = 0; j < 4; ++j) qf[c].u[4 + j] = p[8 + pb + j];
        }
    }

    f32x8 o[8];
    #pragma unroll
    for (int dt = 0; dt < 8; ++dt) o[dt] = zero8();
    float mi[8], li[8];
    #pragma unroll
    for (int r = 0; r < 8; ++r) { mi[r] = -3.0e38f; li[r] = 0.f; }

    for (int s0 = 0; s0 <= q0 + 15; s0 += 32) {
        // ---- scores: S[half] = Q(16x128) . K^T, keys s0+16*half .. +15 ----
        f32x8 S[2] = { zero8(), zero8() };
        #pragma unroll
        for (int half = 0; half < 2; ++half) {
            #pragma unroll
            for (int c = 0; c < 4; ++c) {
                AFrag kf;
                const unsigned int* p = reinterpret_cast<const unsigned int*>(
                    &Kb[((size_t)kvh * T_LEN + s0 + half * 16 + lane15) * DH + c * 32]);
                #pragma unroll
                for (int j = 0; j < 8; ++j) kf.u[j] = p[(j >> 2) * 8 + laneHi * 4 + (j & 3)];
                S[half] = __builtin_amdgcn_wmma_f32_16x16x32_bf16(
                    false, qf[c].v, false, kf.v, (short)0, S[half], false, false);
            }
        }
        // ---- online softmax (rows live in VGPR slot r + 8*laneHi; cols in lanes) ----
        float P0[8], P1[8], alpha[8];
        #pragma unroll
        for (int r = 0; r < 8; ++r) {
            const int row = q0 + r + laneHi * 8;
            float a = S[0][r] * scale; if (s0 + lane15      > row) a = -1.0e30f;
            float b = S[1][r] * scale; if (s0 + 16 + lane15 > row) b = -1.0e30f;
            float mloc = fmaxf(a, b);
            #pragma unroll
            for (int m = 1; m < 16; m <<= 1) mloc = fmaxf(mloc, __shfl_xor(mloc, m, 32));
            const float nm = fmaxf(mi[r], mloc);
            alpha[r] = __expf(mi[r] - nm);
            float pa = __expf(a - nm), pb = __expf(b - nm);
            float rs = pa + pb;
            #pragma unroll
            for (int m = 1; m < 16; m <<= 1) rs += __shfl_xor(rs, m, 32);
            li[r] = li[r] * alpha[r] + rs;
            mi[r] = nm;
            P0[r] = pa; P1[r] = pb;
        }
        #pragma unroll
        for (int dt = 0; dt < 8; ++dt)
            #pragma unroll
            for (int r = 0; r < 8; ++r) o[dt][r] *= alpha[r];

        // ---- P (f32, C-layout) -> LDS -> bf16 A-fragment ----
        __syncthreads();
        #pragma unroll
        for (int r = 0; r < 8; ++r) {
            sP[r + laneHi * 8][lane15]      = to_bf16(P0[r]);
            sP[r + laneHi * 8][16 + lane15] = to_bf16(P1[r]);
        }
        __syncthreads();
        AFrag pf;
        {
            const unsigned int* p = reinterpret_cast<const unsigned int*>(&sP[lane15][0]);
            const int pb = laneHi * 4;
            #pragma unroll
            for (int j = 0; j < 4; ++j) pf.u[j]     = p[pb + j];
            #pragma unroll
            for (int j = 0; j < 4; ++j) pf.u[4 + j] = p[8 + pb + j];
        }
        // ---- PV: o += P(16x32) . V(32xDH), V transposed so keys are contiguous ----
        #pragma unroll
        for (int dt = 0; dt < 8; ++dt) {
            AFrag vf;
            const unsigned int* p = reinterpret_cast<const unsigned int*>(
                &Vt[((size_t)kvh * DH + dt * 16 + lane15) * T_LEN + s0]);
            #pragma unroll
            for (int j = 0; j < 8; ++j) vf.u[j] = p[(j >> 2) * 8 + laneHi * 4 + (j & 3)];
            o[dt] = __builtin_amdgcn_wmma_f32_16x16x32_bf16(
                false, pf.v, false, vf.v, (short)0, o[dt], false, false);
        }
        __syncthreads();
    }

    #pragma unroll
    for (int dt = 0; dt < 8; ++dt)
        #pragma unroll
        for (int r = 0; r < 8; ++r) {
            const int row = q0 + r + laneHi * 8;
            const float v = o[dt][r] / (li[r] + 1e-6f);
            attnb[(size_t)row * (NH * DH) + (size_t)h * DH + dt * 16 + lane15] = to_bf16(v);
        }
}

// =============================================================
// silu(gate) * up -> bf16
// =============================================================
__global__ __launch_bounds__(256)
void silu_mul_kernel(const float* __restrict__ gu, __bf16* __restrict__ out)
{
    long long i = (long long)blockIdx.x * 256 + threadIdx.x;
    if (i >= (long long)T_LEN * HID) return;
    long long t = i / HID, j = i % HID;
    float a = gu[t * NGU + j];
    float b = gu[t * NGU + HID + j];
    out[i] = to_bf16((a / (1.f + __expf(-a))) * b);
}

// =============================================================
// Host orchestration
// =============================================================
extern "C" void kernel_launch(void* const* d_in, const int* in_sizes, int n_in,
                              void* d_out, int out_size, void* d_ws, size_t ws_size,
                              hipStream_t stream)
{
    (void)in_sizes; (void)n_in; (void)out_size; (void)ws_size;
    const float* x       = (const float*)d_in[0];
    const float* ln_w    = (const float*)d_in[1];
    const float* ff_ln_w = (const float*)d_in[2];
    const float* wq      = (const float*)d_in[3];
    const float* wk      = (const float*)d_in[4];
    const float* wv      = (const float*)d_in[5];
    const float* wo      = (const float*)d_in[6];
    const float* wg      = (const float*)d_in[7];
    const float* w1      = (const float*)d_in[8];
    const float* w2      = (const float*)d_in[9];
    float* out = (float*)d_out;

    char* p = (char*)d_ws;
    auto carve = [&](size_t bytes) -> void* {
        void* r = (void*)p;
        p += (bytes + 255) & ~(size_t)255;
        return r;
    };
    __bf16* hB    = (__bf16*)carve((size_t)T_LEN * EMB * 2);
    __bf16* wqkvT = (__bf16*)carve((size_t)NQKV * EMB * 2);   // [NQKV][EMB]
    float*  qkvF  = (float*) carve((size_t)T_LEN * NQKV * 4);
    __bf16* Qb    = (__bf16*)carve((size_t)NH   * T_LEN * DH * 2);
    __bf16* Kb    = (__bf16*)carve((size_t)NKVH * T_LEN * DH * 2);
    __bf16* Vt    = (__bf16*)carve((size_t)NKVH * DH * T_LEN * 2);
    __bf16* attnB = (__bf16*)carve((size_t)T_LEN * EMB * 2);
    __bf16* woT   = (__bf16*)carve((size_t)EMB * EMB * 2);    // [EMB][EMB]
    float*  r1F   = (float*) carve((size_t)T_LEN * EMB * 4);
    __bf16* gB    = (__bf16*)carve((size_t)T_LEN * EMB * 2);
    __bf16* wguT  = (__bf16*)carve((size_t)NGU * EMB * 2);    // [NGU][EMB]
    float*  guF   = (float*) carve((size_t)T_LEN * NGU * 4);
    __bf16* ffB   = (__bf16*)carve((size_t)T_LEN * HID * 2);
    __bf16* w2T   = (__bf16*)carve((size_t)EMB * HID * 2);    // [EMB][HID]

    auto cblk = [](long long n) { return (unsigned)((n + 255) / 256); };

    // 1) h = rmsnorm(x, ln_w)  (bf16)
    rmsnorm_kernel<<<T_LEN, 256, 0, stream>>>(x, ln_w, hB, EMB);

    // 2) fuse + cast weights, transposed to [N][K] for async DMA tiles
    copy_cast_T_kernel<<<cblk((long long)EMB * EMB),  256, 0, stream>>>(wq, wqkvT, (long long)EMB * EMB,  EMB,       EMB, 0);
    copy_cast_T_kernel<<<cblk((long long)EMB * 1024), 256, 0, stream>>>(wk, wqkvT, (long long)EMB * 1024, NKVH * DH, EMB, EMB);
    copy_cast_T_kernel<<<cblk((long long)EMB * 1024), 256, 0, stream>>>(wv, wqkvT, (long long)EMB * 1024, NKVH * DH, EMB, EMB + NKVH * DH);
    copy_cast_T_kernel<<<cblk((long long)EMB * EMB),  256, 0, stream>>>(wo, woT,   (long long)EMB * EMB,  EMB,       EMB, 0);
    copy_cast_T_kernel<<<cblk((long long)EMB * HID),  256, 0, stream>>>(wg, wguT,  (long long)EMB * HID,  HID,       EMB, 0);
    copy_cast_T_kernel<<<cblk((long long)EMB * HID),  256, 0, stream>>>(w1, wguT,  (long long)EMB * HID,  HID,       EMB, HID);
    copy_cast_T_kernel<<<cblk((long long)HID * EMB),  256, 0, stream>>>(w2, w2T,   (long long)HID * EMB,  EMB,       HID, 0);

    // 3) qkv = h @ [wq|wk|wv]
    gemm_bf16_wmma<<<dim3(NQKV / TILE_N, T_LEN / TILE_M), 256, 0, stream>>>(
        hB, wqkvT, nullptr, qkvF, T_LEN, NQKV, EMB);

    // 4) RoPE + bf16 layouts
    rope_kernel<<<T_LEN, 256, 0, stream>>>(qkvF, Qb, Kb, Vt);

    // 5) attention (flash, WMMA)
    attn_kernel<<<dim3(T_LEN / 16, NH), 32, 0, stream>>>(Qb, Kb, Vt, attnB);

    // 6) r1 = x + attn @ wo
    gemm_bf16_wmma<<<dim3(EMB / TILE_N, T_LEN / TILE_M), 256, 0, stream>>>(
        attnB, woT, x, r1F, T_LEN, EMB, EMB);

    // 7) g = rmsnorm(r1, ff_ln_w)
    rmsnorm_kernel<<<T_LEN, 256, 0, stream>>>(r1F, ff_ln_w, gB, EMB);

    // 8) gu = g @ [wg|w1]
    gemm_bf16_wmma<<<dim3(NGU / TILE_N, T_LEN / TILE_M), 256, 0, stream>>>(
        gB, wguT, nullptr, guF, T_LEN, NGU, EMB);

    // 9) ff_in = silu(gate) * up
    silu_mul_kernel<<<cblk((long long)T_LEN * HID), 256, 0, stream>>>(guF, ffB);

    // 10) out = r1 + ff_in @ w2
    gemm_bf16_wmma<<<dim3(EMB / TILE_N, T_LEN / TILE_M), 256, 0, stream>>>(
        ffB, w2T, r1F, out, T_LEN, EMB, HID);
}
